// SSDBoxHead_36696200577242
// MI455X (gfx1250) — matrix operand order repaired and checked by reading, not compiled, not run on hardware
//
#include <hip/hip_runtime.h>

typedef __attribute__((ext_vector_type(16))) _Float16 v16h;
typedef __attribute__((ext_vector_type(8)))  float    v8f;

#define WAVES        4
#define NUM_CLASSES  21
#define NUM_PRIORS   8732
#define BATCH        32

// ---------------------------------------------------------------------------
// Weight prep: f32 (O,C,3,3) cls+reg fused -> f16 B-fragments in WMMA order.
// K ordering is (tap q = ky*3+kx outer, channel c inner):  kn = q*C + c.
// B layout (16x16x32 f16, wave32): lane L holds column n = L&15,
// element j holds k = 16*(L>>4) + j within the 32-wide k-tile.
// Global layout: bfrag[((t*NT + u)*32 + lane)*16 + j]
// ---------------------------------------------------------------------------
__global__ void ssd_prep_weights(const float* __restrict__ cls_w,
                                 const float* __restrict__ reg_w,
                                 _Float16* __restrict__ bfrag,
                                 int C, int A, int NT, int total)
{
    int idx = blockIdx.x * blockDim.x + threadIdx.x;
    if (idx >= total) return;
    int j    = idx & 15;
    int lane = (idx >> 4) & 31;
    int u    = (idx >> 9) % NT;
    int t    = idx / (512 * NT);
    int n    = u * 16 + (lane & 15);
    int kn   = t * 32 + ((lane >> 4) << 4) + j;   // kn = q*C + c
    int q    = kn / C;
    int c    = kn - q * C;
    float w  = 0.0f;
    int ncls = A * NUM_CLASSES;
    if (n < ncls)          w = cls_w[(size_t)n * (C * 9) + c * 9 + q];
    else if (n < A * 25)   w = reg_w[(size_t)(n - ncls) * (C * 9) + c * 9 + q];
    bfrag[idx] = (_Float16)w;
}

// ---------------------------------------------------------------------------
// Implicit-GEMM conv head + fused softmax/box-decode epilogue.
// - One wave per 16-row M tile; NT f32 accumulators cover all N.
// - A gather: channel-strided loads with compile-time immediate offsets,
//   software-pipelined one k-tile ahead.
// - B path: block-cooperative async global->LDS staging (ASYNCcnt), double
//   buffered, so the copy of k-tile t+1 overlaps the WMMAs of k-tile t;
//   fragments are then read with low-latency ds_load_b128.
// ---------------------------------------------------------------------------
template<int C, int F, int A, int NT>
__global__ __launch_bounds__(WAVES * 32)
void ssd_head_gemm(const float*    __restrict__ feat,
                   const _Float16* __restrict__ bfrag,
                   const float*    __restrict__ cls_b,
                   const float*    __restrict__ reg_b,
                   const float*    __restrict__ priors,
                   float*          __restrict__ scores,   // [B, 8732, 21]
                   float*          __restrict__ boxes,    // [B, 8732, 4]
                   int poff)
{
    constexpr int FF     = F * F;
    constexpr int Mtiles = (BATCH * FF) / 16;
    constexpr int cpt    = C >> 5;        // channel k-tiles per tap
    constexpr int ktiles = 9 * cpt;
    constexpr int STRIDE = NT * 16 + 1;
    constexpr int SLAB   = NT * 512;      // halfs per B k-tile slab
    constexpr int SLABB  = NT * 1024;     // bytes per slab

    __shared__ alignas(16) _Float16 bstage[2 * SLAB];
    __shared__ float tile[WAVES * 16 * STRIDE];

    const int tid   = threadIdx.x;
    const int lane  = tid & 31;
    const int wave  = tid >> 5;
    const int mtileRaw = blockIdx.x * WAVES + wave;
    const bool active  = (mtileRaw < Mtiles);
    const int mtile    = active ? mtileRaw : 0;   // clamp: tail waves still
                                                  // run loop + barriers
    const int m0 = mtile * 16;

    // --- A-fragment addressing (contiguous k order: element j is k = 16*(lane>>4) + j) ---
    const int rowA = m0 + (lane & 15);
    const int bA   = rowA / FF;
    const int rA   = rowA - bA * FF;
    const int yA   = rA / F;
    const int xA   = rA - yA * F;
    const int b8   = (lane >> 4) * 16;
    const float* fbaseL = feat + (size_t)bA * C * FF + (size_t)b8 * FF;

    auto buildA = [&](int t) -> v16h {
        const int q   = t / cpt;
        const int ct  = t - q * cpt;
        const int ky  = q / 3;
        const int kx  = q - 3 * ky;
        const int iy  = yA + ky - 1;
        const int ix  = xA + kx - 1;
        const bool inb = ((unsigned)iy < (unsigned)F) && ((unsigned)ix < (unsigned)F);
        const float* sp = fbaseL + (size_t)(ct * 32) * FF + (inb ? (iy * F + ix) : 0);
        v16h a;
#pragma unroll
        for (int j = 0; j < 16; ++j) {
            const int cj = j;                          // + b8 folded into fbaseL
            float v = sp[cj * FF];                     // immediate-offset load
            a[j] = (_Float16)(inb ? v : 0.0f);
        }
        return a;
    };

    // --- async B staging: whole block copies one SLAB, NT x b64 per thread ---
    const uint32_t bstageAddr = (uint32_t)(uintptr_t)&bstage[0];
    auto stage = [&](int t) {
        const uint64_t gbase = (uint64_t)(uintptr_t)bfrag + (uint64_t)t * SLABB;
        const uint32_t voff  = (uint32_t)tid * 8u;
        const uint32_t laddr = bstageAddr + (uint32_t)((t & 1) * SLABB) + voff;
#pragma unroll
        for (int rdx = 0; rdx < NT; ++rdx)
            asm volatile("global_load_async_to_lds_b64 %0, %1, %2 offset:%3"
                         :: "v"(laddr + rdx * 1024), "v"(voff + rdx * 1024 - rdx * 1024),
                            "s"(gbase), "i"(rdx * 1024)
                         : "memory");
    };

    v8f acc[NT] = {};

    stage(0);
    asm volatile("s_wait_asynccnt 0x0" ::: "memory");
    __syncthreads();

    v16h afrag = buildA(0);

    for (int t = 0; t < ktiles; ++t) {
        if (t + 1 < ktiles) stage(t + 1);          // overlaps WMMAs below

        const _Float16* bbase = &bstage[(t & 1) * SLAB] + (size_t)lane * 16;

        // build next A fragment (its global loads overlap too)
        v16h an = (t + 1 < ktiles) ? buildA(t + 1) : afrag;

#pragma unroll
        for (int nt = 0; nt < NT; ++nt) {
            v16h bfr = *(const v16h*)(bbase + nt * 512);   // ds_load_b128 x2
            acc[nt] = __builtin_amdgcn_wmma_f32_16x16x32_f16(
                false, afrag, false, bfr, (short)0, acc[nt], false, false);
        }
        afrag = an;

        asm volatile("s_wait_asynccnt 0x0" ::: "memory");
        __syncthreads();
    }

    // --- spill C tiles to LDS (C layout: lane&15 = n, lane>>4 ? M=8+j : M=j) ---
    float* tw = tile + wave * 16 * STRIDE;
    const int nbase = lane & 15;
    const int mbase = (lane >> 4) * 8;
#pragma unroll
    for (int nt = 0; nt < NT; ++nt)
#pragma unroll
        for (int j = 0; j < 8; ++j)
            tw[(mbase + j) * STRIDE + nt * 16 + nbase] = acc[nt][j];

    asm volatile("s_wait_dscnt 0x0" ::: "memory");

    if (!active) return;

    // --- fused epilogue: lane handles row (lane&15), anchors a%2 == lane>>4 ---
    const int r  = lane & 15;
    const int m  = m0 + r;
    const int b  = m / FF;
    const int rm = m - b * FF;
    const int y  = rm / F;
    const int x  = rm - y * F;
    const float* trow = tw + r * STRIDE;

    for (int a = (lane >> 4); a < A; a += 2) {
        const int p = poff + (y * F + x) * A + a;

        float lg[NUM_CLASSES];
        float mx = -1e30f;
#pragma unroll
        for (int cc = 0; cc < NUM_CLASSES; ++cc) {
            float v = trow[a * NUM_CLASSES + cc] + cls_b[a * NUM_CLASSES + cc];
            lg[cc] = v;
            mx = fmaxf(mx, v);
        }
        float s = 0.0f;
#pragma unroll
        for (int cc = 0; cc < NUM_CLASSES; ++cc) { lg[cc] = __expf(lg[cc] - mx); s += lg[cc]; }
        float inv = 1.0f / s;
        float* so = scores + ((size_t)b * NUM_PRIORS + p) * NUM_CLASSES;
#pragma unroll
        for (int cc = 0; cc < NUM_CLASSES; ++cc) so[cc] = lg[cc] * inv;

        float l0 = trow[A * NUM_CLASSES + a * 4 + 0] + reg_b[a * 4 + 0];
        float l1 = trow[A * NUM_CLASSES + a * 4 + 1] + reg_b[a * 4 + 1];
        float l2 = trow[A * NUM_CLASSES + a * 4 + 2] + reg_b[a * 4 + 2];
        float l3 = trow[A * NUM_CLASSES + a * 4 + 3] + reg_b[a * 4 + 3];
        float px = priors[p * 4 + 0], py = priors[p * 4 + 1];
        float pw = priors[p * 4 + 2], ph = priors[p * 4 + 3];
        float cx = l0 * 0.1f * pw + px;
        float cy = l1 * 0.1f * ph + py;
        float w  = __expf(l2 * 0.2f) * pw;
        float h  = __expf(l3 * 0.2f) * ph;
        float* bo = boxes + ((size_t)b * NUM_PRIORS + p) * 4;
        bo[0] = cx - 0.5f * w;
        bo[1] = cy - 0.5f * h;
        bo[2] = cx + 0.5f * w;
        bo[3] = cy + 0.5f * h;
    }
}

// ---------------------------------------------------------------------------
extern "C" void kernel_launch(void* const* d_in, const int* in_sizes, int n_in,
                              void* d_out, int out_size, void* d_ws, size_t ws_size,
                              hipStream_t stream)
{
    (void)in_sizes; (void)n_in; (void)out_size; (void)ws_size;

    const float* feat[6];
    for (int l = 0; l < 6; ++l) feat[l] = (const float*)d_in[l];
    const float* priors = (const float*)d_in[6];
    const float *cls_w[6], *cls_bp[6], *reg_w[6], *reg_bp[6];
    for (int l = 0; l < 6; ++l) {
        cls_w[l]  = (const float*)d_in[7  + 4 * l];
        cls_bp[l] = (const float*)d_in[8  + 4 * l];
        reg_w[l]  = (const float*)d_in[9  + 4 * l];
        reg_bp[l] = (const float*)d_in[10 + 4 * l];
    }

    float* scores = (float*)d_out;
    float* boxes  = scores + (size_t)BATCH * NUM_PRIORS * NUM_CLASSES;
    _Float16* bfrag = (_Float16*)d_ws;

    const int Cs[6]  = {512, 1024, 512, 256, 256, 256};
    const int Fs[6]  = {38, 19, 10, 5, 3, 1};
    const int As[6]  = {4, 6, 6, 6, 4, 4};
    const int NTs[6] = {7, 10, 10, 10, 7, 7};

    size_t boff[6];
    int poff[6];
    {
        size_t off = 0; int po = 0;
        for (int l = 0; l < 6; ++l) {
            poff[l] = po;  po += Fs[l] * Fs[l] * As[l];
            boff[l] = off; off += (size_t)Cs[l] * 9 * NTs[l] * 16;
        }
    }

    // Phase 1: weight fragment prep (f32 -> f16 WMMA B-fragment order)
    for (int l = 0; l < 6; ++l) {
        int total = Cs[l] * 9 * NTs[l] * 16;
        ssd_prep_weights<<<(total + 255) / 256, 256, 0, stream>>>(
            cls_w[l], reg_w[l], bfrag + boff[l], Cs[l], As[l], NTs[l], total);
    }

    // Phase 2: implicit-GEMM conv + fused softmax/box-decode
    {
        int bl;
        bl = ((BATCH * 38 * 38 / 16) + WAVES - 1) / WAVES;
        ssd_head_gemm<512, 38, 4, 7><<<bl, WAVES * 32, 0, stream>>>(
            feat[0], bfrag + boff[0], cls_bp[0], reg_bp[0], priors, scores, boxes, poff[0]);
        bl = ((BATCH * 19 * 19 / 16) + WAVES - 1) / WAVES;
        ssd_head_gemm<1024, 19, 6, 10><<<bl, WAVES * 32, 0, stream>>>(
            feat[1], bfrag + boff[1], cls_bp[1], reg_bp[1], priors, scores, boxes, poff[1]);
        bl = ((BATCH * 10 * 10 / 16) + WAVES - 1) / WAVES;
        ssd_head_gemm<512, 10, 6, 10><<<bl, WAVES * 32, 0, stream>>>(
            feat[2], bfrag + boff[2], cls_bp[2], reg_bp[2], priors, scores, boxes, poff[2]);
        bl = ((BATCH * 5 * 5 / 16) + WAVES - 1) / WAVES;
        ssd_head_gemm<256, 5, 6, 10><<<bl, WAVES * 32, 0, stream>>>(
            feat[3], bfrag + boff[3], cls_bp[3], reg_bp[3], priors, scores, boxes, poff[3]);
        bl = ((BATCH * 3 * 3 / 16) + WAVES - 1) / WAVES;
        ssd_head_gemm<256, 3, 4, 7><<<bl, WAVES * 32, 0, stream>>>(
            feat[4], bfrag + boff[4], cls_bp[4], reg_bp[4], priors, scores, boxes, poff[4]);
        bl = ((BATCH * 1 * 1 / 16) + WAVES - 1) / WAVES;
        ssd_head_gemm<256, 1, 4, 7><<<bl, WAVES * 32, 0, stream>>>(
            feat[5], bfrag + boff[5], cls_bp[5], reg_bp[5], priors, scores, boxes, poff[5]);
    }
}
